// GCNEncoder_10608569221206
// MI455X (gfx1250) — matrix operand order repaired and checked by reading, not compiled, not run on hardware
//
#include <hip/hip_runtime.h>
#include <hip/hip_bf16.h>

typedef __attribute__((ext_vector_type(16))) __bf16 v16bf;
typedef __attribute__((ext_vector_type(8)))  __bf16 v8bf;
typedef __attribute__((ext_vector_type(4)))  __bf16 v4bf;
typedef __attribute__((ext_vector_type(8)))  float  v8f;

#define N_NODES 100000
#define N_EDGES 1600000
#define IN_CH   128
#define H1      256
#define H2      128
#define OUT_CH  64

// ---------------- degree / normalization ----------------

__global__ void deg_init_kernel(float* __restrict__ deg, int n) {
    int i = blockIdx.x * blockDim.x + threadIdx.x;
    if (i < n) deg[i] = 1.0f;  // self-loop contributes 1
}

__global__ void deg_accum_kernel(const int* __restrict__ ei, float* __restrict__ deg, int E) {
    int i = blockIdx.x * blockDim.x + threadIdx.x;
    if (i < E) atomicAdd(&deg[ei[E + i]], 1.0f);  // dst row of edge_index
}

__global__ void deg_rsqrt_kernel(float* __restrict__ d, int n) {
    int i = blockIdx.x * blockDim.x + threadIdx.x;
    if (i < n) d[i] = rsqrtf(d[i]);  // deg >= 1 always
}

// norm[e] = dinv[src]*dinv[dst] (computed once, reused by all 3 layers)
__global__ void edge_norm_kernel(const int* __restrict__ ei, const float* __restrict__ dinv,
                                 float* __restrict__ norm, int E) {
    int e = blockIdx.x * blockDim.x + threadIdx.x;
    if (e < E) norm[e] = dinv[ei[e]] * dinv[ei[E + e]];
}

// ---------------- weight pack: f32 row-major -> bf16 WMMA-fragment layout ----------------
// Packed index t = ((nt*KT + kt)*32 + lane)*16 + j  holds
//   bf16( W[ (kt*32 + (lane<16 ? 0 : 8) + (j<8 ? j : j+8)) * N + nt*16 + (lane&15) ] )
__global__ void packB_kernel(const float* __restrict__ W, __bf16* __restrict__ out,
                             int N, int KT) {
    int t = blockIdx.x * blockDim.x + threadIdx.x;
    int total = KT * 32 * N;  // == K*N
    if (t >= total) return;
    int j  = t & 15;
    int l  = (t >> 4) & 31;
    int g  = t >> 9;          // nt*KT + kt
    int kt = g % KT;
    int nt = g / KT;
    int k   = kt * 32 + ((l < 16) ? 0 : 8) + (j < 8 ? j : j + 8);
    int col = nt * 16 + (l & 15);
    out[t] = (__bf16)W[(size_t)k * N + col];
}

// ---------------- activation convert f32 -> bf16 row-major (optional ReLU) ----------------
__global__ void cvtA_kernel(const float* __restrict__ in, __bf16* __restrict__ out,
                            int n4, int doRelu) {
    int t = blockIdx.x * blockDim.x + threadIdx.x;
    if (t >= n4) return;
    float4 v = ((const float4*)in)[t];
    if (doRelu) {
        v.x = fmaxf(v.x, 0.0f); v.y = fmaxf(v.y, 0.0f);
        v.z = fmaxf(v.z, 0.0f); v.w = fmaxf(v.w, 0.0f);
    }
    v4bf o;
    o[0] = (__bf16)v.x; o[1] = (__bf16)v.y; o[2] = (__bf16)v.z; o[3] = (__bf16)v.w;
    ((v4bf*)out)[t] = o;
}

// ---------------- fused WMMA GEMM + self-loop/bias epilogue ----------------
// P[M,N] = A[M,K] @ W[K,N]          (pre-aggregation features, read by edge scatter)
// Q[M,N] = P * dinv[row]^2 + bias   (self-loop + bias init of the aggregation output)
// Block = 4 waves. Each wave: 4 row tiles x NT col tiles. K fully unrolled.
// Fragment layouts per CDNA5 ISA 7.12.2 (wave32).
template <int K, int NT>
__global__ __launch_bounds__(128)
void gemm_fused_kernel(const __bf16* __restrict__ A, const __bf16* __restrict__ Bp,
                       float* __restrict__ P, float* __restrict__ Q,
                       const float* __restrict__ dinv, const float* __restrict__ bias,
                       int Mtiles, int N) {
    constexpr int KT = K / 32;
    const int lane = threadIdx.x & 31;
    const int wave = threadIdx.x >> 5;
    const int half = lane >> 4;    // 0 or 1
    const int l15  = lane & 15;

    const int ntBase = (blockIdx.y * 4 + wave) * NT;   // first 16-col tile of this wave
    const int mt0    = blockIdx.x * 4;

    bool valid[4];
    const __bf16* aRow[4];
#pragma unroll
    for (int mt = 0; mt < 4; ++mt) {
        valid[mt] = (mt0 + mt) < Mtiles;
        int rt = valid[mt] ? (mt0 + mt) : (Mtiles - 1);   // clamp: keep loads in-bounds
        aRow[mt] = A + (size_t)(rt * 16 + l15) * K + half * 8;
    }
    const __bf16* bPtr[NT];
#pragma unroll
    for (int n = 0; n < NT; ++n)
        bPtr[n] = Bp + (size_t)(ntBase + n) * KT * 512 + lane * 16;

    v8f acc[4][NT];
#pragma unroll
    for (int mt = 0; mt < 4; ++mt)
#pragma unroll
        for (int n = 0; n < NT; ++n)
            acc[mt][n] = (v8f){};

#pragma unroll
    for (int kt = 0; kt < KT; ++kt) {
        union { v16bf v; v8bf h[2]; } bfr[NT];
#pragma unroll
        for (int n = 0; n < NT; ++n) {
            bfr[n].h[0] = *(const v8bf*)(bPtr[n] + kt * 512);
            bfr[n].h[1] = *(const v8bf*)(bPtr[n] + kt * 512 + 8);
        }
#pragma unroll
        for (int mt = 0; mt < 4; ++mt) {
            union { v16bf v; v8bf h[2]; } afr;
            const __bf16* ap = aRow[mt] + kt * 32;
            afr.h[0] = *(const v8bf*)(ap);       // K = aKlo .. aKlo+7
            afr.h[1] = *(const v8bf*)(ap + 16);  // K = aKlo+16 .. aKlo+23
#pragma unroll
            for (int n = 0; n < NT; ++n)
                acc[mt][n] = __builtin_amdgcn_wmma_f32_16x16x32_bf16(
                    false, afr.v, false, bfr[n].v, (short)0, acc[mt][n], false, false);
        }
    }

    // epilogue: P = acc ; Q = acc * dinv[row]^2 + bias[col]
    float bv[NT];
#pragma unroll
    for (int n = 0; n < NT; ++n)
        bv[n] = bias[(ntBase + n) * 16 + l15];

#pragma unroll
    for (int mt = 0; mt < 4; ++mt) {
        if (!valid[mt]) continue;
        const int rbase = (mt0 + mt) * 16 + half * 8;
        float dv[8];
#pragma unroll
        for (int ri = 0; ri < 8; ++ri) {
            float s = dinv[rbase + ri];
            dv[ri] = s * s;
        }
#pragma unroll
        for (int n = 0; n < NT; ++n) {
            const int c = (ntBase + n) * 16 + l15;
            float* Pp = P + (size_t)rbase * N + c;
            float* Qp = Q + (size_t)rbase * N + c;
#pragma unroll
            for (int ri = 0; ri < 8; ++ri) {
                float v = acc[mt][n][ri];
                Pp[(size_t)ri * N] = v;
                Qp[(size_t)ri * N] = v * dv[ri] + bv[n];
            }
        }
    }
}

// ---------------- aggregation: edge scatter ----------------
// for each edge e: out[dst,c] += h[src,c] * norm[e]
__global__ void agg_edges_kernel(const float* __restrict__ h, const float* __restrict__ norm,
                                 const int* __restrict__ ei, float* __restrict__ out,
                                 int E, int C, int cpeShift) {
    int t = blockIdx.x * blockDim.x + threadIdx.x;
    int total = E << cpeShift;           // E * (C/4), max 102.4M < 2^31
    if (t >= total) return;
    int e  = t >> cpeShift;
    int c4 = (t & ((1 << cpeShift) - 1)) << 2;
    int s = ei[e];
    int d = ei[E + e];
    float nrm = norm[e];
    const float4 hv = *(const float4*)(h + (size_t)s * C + c4);
    float* op = out + (size_t)d * C + c4;
    atomicAdd(op + 0, hv.x * nrm);
    atomicAdd(op + 1, hv.y * nrm);
    atomicAdd(op + 2, hv.z * nrm);
    atomicAdd(op + 3, hv.w * nrm);
}

// ---------------- launcher ----------------

static inline int ceil_div(int a, int b) { return (a + b - 1) / b; }

extern "C" void kernel_launch(void* const* d_in, const int* in_sizes, int n_in,
                              void* d_out, int out_size, void* d_ws, size_t ws_size,
                              hipStream_t stream) {
    const float* x   = (const float*)d_in[0];
    const int*   ei  = (const int*)d_in[1];
    const float* W1  = (const float*)d_in[2];
    const float* b1  = (const float*)d_in[3];
    const float* W2  = (const float*)d_in[4];
    const float* b2  = (const float*)d_in[5];
    const float* W3  = (const float*)d_in[6];
    const float* b3  = (const float*)d_in[7];
    float* out = (float*)d_out;

    // workspace layout (offsets 256B aligned)
    char* ws = (char*)d_ws;
    const size_t szPQ = (size_t)N_NODES * H1 * sizeof(float);    // 102,400,000 B
    const size_t szAb = (size_t)N_NODES * H1 * sizeof(__bf16);   //  51,200,000 B
    const size_t szDi = (((size_t)N_NODES * 4 + 255) & ~(size_t)255);
    const size_t szNm = (size_t)N_EDGES * sizeof(float);         //   6,400,000 B
    float*  P    = (float*)(ws);
    float*  Q    = (float*)(ws + szPQ);
    __bf16* Abf  = (__bf16*)(ws + 2 * szPQ);
    float*  dinv = (float*)(ws + 2 * szPQ + szAb);
    float*  norm = (float*)(ws + 2 * szPQ + szAb + szDi);
    char*   wB   = ws + 2 * szPQ + szAb + szDi + szNm;
    __bf16* w1p  = (__bf16*)(wB);
    __bf16* w2p  = (__bf16*)(wB + 65536);
    __bf16* w3p  = (__bf16*)(wB + 131072);

    const int TB = 256;
    const int Mtiles = N_NODES / 16;          // 6250
    const int gx = ceil_div(Mtiles, 4);       // 1563

    // 1) degree + dinv + per-edge norm
    deg_init_kernel<<<ceil_div(N_NODES, TB), TB, 0, stream>>>(dinv, N_NODES);
    deg_accum_kernel<<<ceil_div(N_EDGES, TB), TB, 0, stream>>>(ei, dinv, N_EDGES);
    deg_rsqrt_kernel<<<ceil_div(N_NODES, TB), TB, 0, stream>>>(dinv, N_NODES);
    edge_norm_kernel<<<ceil_div(N_EDGES, TB), TB, 0, stream>>>(ei, dinv, norm, N_EDGES);

    // 2) weights -> packed bf16 fragments
    packB_kernel<<<ceil_div(IN_CH * H1, TB), TB, 0, stream>>>(W1, w1p, H1, IN_CH / 32);
    packB_kernel<<<ceil_div(H1 * H2, TB), TB, 0, stream>>>(W2, w2p, H2, H1 / 32);
    packB_kernel<<<ceil_div(H2 * OUT_CH, TB), TB, 0, stream>>>(W3, w3p, OUT_CH, H2 / 32);

    // ---- layer 1: P = x @ W1 ; Q = P*dinv^2 + b1 ; Q += scatter(P) ----
    cvtA_kernel<<<ceil_div(N_NODES * IN_CH / 4, TB), TB, 0, stream>>>(x, Abf, N_NODES * IN_CH / 4, 0);
    gemm_fused_kernel<IN_CH, 2><<<dim3(gx, H1 / 128), 128, 0, stream>>>(Abf, w1p, P, Q, dinv, b1, Mtiles, H1);
    agg_edges_kernel<<<ceil_div(N_EDGES / 4 * H1, TB), TB, 0, stream>>>(P, norm, ei, Q, N_EDGES, H1, 6);

    // ---- layer 2: P = Q @ W2 ; Q = P*dinv^2 + b2 ; Q += scatter(P) ----
    cvtA_kernel<<<ceil_div(N_NODES * H1 / 4, TB), TB, 0, stream>>>(Q, Abf, N_NODES * H1 / 4, 0);
    gemm_fused_kernel<H1, 2><<<dim3(gx, H2 / 128), 128, 0, stream>>>(Abf, w2p, P, Q, dinv, b2, Mtiles, H2);
    agg_edges_kernel<<<ceil_div(N_EDGES / 4 * H2, TB), TB, 0, stream>>>(P, norm, ei, Q, N_EDGES, H2, 5);

    // ---- layer 3: P = relu(Q) @ W3 ; out = P*dinv^2 + b3 ; out += scatter(P) ----
    cvtA_kernel<<<ceil_div(N_NODES * H2 / 4, TB), TB, 0, stream>>>(Q, Abf, N_NODES * H2 / 4, 1);
    gemm_fused_kernel<H2, 1><<<dim3(gx, OUT_CH / 64), 128, 0, stream>>>(Abf, w3p, P, out, dinv, b3, Mtiles, OUT_CH);
    agg_edges_kernel<<<ceil_div(N_EDGES / 4 * OUT_CH, TB), TB, 0, stream>>>(P, norm, ei, out, N_EDGES, OUT_CH, 4);
}
